// HyperEncoder_66099546685836
// MI455X (gfx1250) — matrix operand order repaired and checked by reading, not compiled, not run on hardware
//
#include <hip/hip_runtime.h>
#include <stdint.h>

#define S_    256
#define B_    64
#define E_    512
#define H_    512
#define L_    64
#define FOURH 2048   // 4*H
#define KTOT  1024   // E + H
#define XSTRIDE 520  // 512 + 8 shorts padding (conflict-free ds_read_b128, TDM pad 4 dwords)

typedef __bf16 v16bf __attribute__((ext_vector_type(16)));
typedef float  v8f   __attribute__((ext_vector_type(8)));
typedef unsigned int u32x4 __attribute__((ext_vector_type(4)));
typedef int          i32x4 __attribute__((ext_vector_type(4)));
typedef int          i32x8 __attribute__((ext_vector_type(8)));

union Frag { v16bf v; uint4 q[2]; };

#if __has_builtin(__builtin_amdgcn_tensor_load_to_lds)
#define HAVE_TDM 1
#else
#define HAVE_TDM 0
#endif

__device__ __forceinline__ unsigned short f2bf(float f) {
  unsigned int u = __float_as_uint(f);
  unsigned int r = u + 0x7FFFu + ((u >> 16) & 1u);   // round-to-nearest-even
  return (unsigned short)(r >> 16);
}

__device__ __forceinline__ float sigm(float x)   { return 1.0f / (1.0f + __expf(-x)); }
__device__ __forceinline__ float tanh_f(float x) { float e = __expf(2.0f * x); return (e - 1.0f) / (e + 1.0f); }

#if HAVE_TDM
// TDM: DMA a 16-row x 512-bf16 tile (row stride 512 elems) into LDS with
// 4-dword padding every 256 dwords -> LDS row stride 520 shorts.
__device__ __forceinline__ void issue_tdm_x(const unsigned short* gsrc,
                                            unsigned int lds_byte_off) {
  unsigned long long ga = (unsigned long long)(size_t)gsrc;
  u32x4 g0 = { 1u,                                         // count=1 (valid D#)
               lds_byte_off,                               // lds_addr
               (unsigned int)(ga & 0xFFFFFFFFu),           // global_addr[31:0]
               (unsigned int)((ga >> 32) & 0x01FFFFFFu) | 0x80000000u }; // [56:32] | type=2
  i32x8 g1 = { (int)((1u << 16) | (1u << 20) | (7u << 22) | (3u << 25)),
               //    data_size=2B  pad_enable  interval=256dw  amount=4dw
               (int)(512u << 16),    // tensor_dim0 = 512 (bits 79:48 -> low16 here)
               (int)(16u  << 16),    // tensor_dim0 hi=0 | tensor_dim1 = 16
               (int)(512u << 16),    // tensor_dim1 hi=0 | tile_dim0 = 512
               16,                   // tile_dim1 = 16, tile_dim2 = 0
               512,                  // tensor_dim0_stride low32 = 512 elems
               (int)(512u << 16),    // stride0 hi=0 | tensor_dim1_stride low16 = 512
               0 };                  // tensor_dim1_stride hi
  i32x4 z4 = { 0, 0, 0, 0 };
#if defined(__clang_major__) && (__clang_major__ >= 23)
  i32x8 z8 = { 0, 0, 0, 0, 0, 0, 0, 0 };
  __builtin_amdgcn_tensor_load_to_lds(g0, g1, z4, z4, z8, 0);
#else
  __builtin_amdgcn_tensor_load_to_lds(g0, g1, z4, z4, 0);
#endif
}
#endif

// ---------------------------------------------------------------------------
// K1: embedding gather + f32 -> bf16 convert.
// ---------------------------------------------------------------------------
__global__ void embed_bf16_kernel(const int* __restrict__ x,
                                  const float* __restrict__ table,
                                  unsigned short* __restrict__ embBF) {
  int idx = blockIdx.x * blockDim.x + threadIdx.x;  // over S*B*E
  int e  = idx & (E_ - 1);
  int sb = idx >> 9;
  int tok = x[sb];
  embBF[idx] = f2bf(table[(size_t)tok * E_ + e]);
}

// ---------------------------------------------------------------------------
// K2: hypernet weight generation (~1 GB HBM-bound streaming pass).
// Stores bf16 in WMMA B-fragment-swizzled order.
// ---------------------------------------------------------------------------
__global__ void genw_kernel(const float* __restrict__ lang_g,
    const float* __restrict__ wih_w,   const float* __restrict__ wih_b,
    const float* __restrict__ whh_w,   const float* __restrict__ whh_b,
    const float* __restrict__ wih_r_w, const float* __restrict__ wih_r_b,
    const float* __restrict__ whh_r_w, const float* __restrict__ whh_r_b,
    unsigned short* __restrict__ Wfrag) {
  __shared__ float lang[L_];
  int tid = threadIdx.x;
  if (tid < L_) lang[tid] = lang_g[tid];
  __syncthreads();

  int idx = blockIdx.x * blockDim.x + tid;   // 2 * 2048 * 1024
  int k   = idx & (KTOT - 1);
  int n   = (idx >> 10) & (FOURH - 1);
  int dir = idx >> 21;

  const float* wsrc; const float* bsrc; int ridx;
  if (k < E_) { wsrc = dir ? wih_r_w : wih_w; bsrc = dir ? wih_r_b : wih_b; ridx = n * E_ + k; }
  else        { wsrc = dir ? whh_r_w : whh_w; bsrc = dir ? whh_r_b : whh_b; ridx = n * H_ + (k - E_); }

  const float4* wr = (const float4*)(wsrc + (size_t)ridx * L_);
  float acc = bsrc[ridx];
#pragma unroll
  for (int i = 0; i < L_ / 4; ++i) {
    float4 w4 = wr[i];
    acc += w4.x * lang[4*i] + w4.y * lang[4*i+1] + w4.z * lang[4*i+2] + w4.w * lang[4*i+3];
  }

  int ntile = n >> 4, nl = n & 15;
  int kb = k >> 5,  kr = k & 31;
  int half = kr >> 4, pos = kr & 15;
  size_t addr = ((size_t)dir * FOURH * KTOT) +
                (((size_t)(ntile * 32 + kb) * 32 + half * 16 + nl) * 16 + pos);
  Wfrag[addr] = f2bf(acc);
}

// ---------------------------------------------------------------------------
// K3: combined gate biases b = gen(bih) + gen(bhh), per direction.
// ---------------------------------------------------------------------------
__global__ void genb_kernel(const float* __restrict__ lang_g,
    const float* __restrict__ bih_w,   const float* __restrict__ bih_b,
    const float* __restrict__ bhh_w,   const float* __restrict__ bhh_b,
    const float* __restrict__ bih_r_w, const float* __restrict__ bih_r_b,
    const float* __restrict__ bhh_r_w, const float* __restrict__ bhh_r_b,
    float* __restrict__ bias) {
  __shared__ float lang[L_];
  int tid = threadIdx.x;
  if (tid < L_) lang[tid] = lang_g[tid];
  __syncthreads();
  int idx = blockIdx.x * blockDim.x + tid;  // 2 * 2048
  int n = idx & (FOURH - 1);
  int dir = idx >> 11;
  const float* w1 = dir ? bih_r_w : bih_w; const float* b1 = dir ? bih_r_b : bih_b;
  const float* w2 = dir ? bhh_r_w : bhh_w; const float* b2 = dir ? bhh_r_b : bhh_b;
  float acc = b1[n] + b2[n];
  for (int i = 0; i < L_; ++i) acc += (w1[n * L_ + i] + w2[n * L_ + i]) * lang[i];
  bias[idx] = acc;
}

// ---------------------------------------------------------------------------
// K4: persistent bidirectional LSTM. 8 independent workgroups (dir x batch16).
// Per step: TDM prefetches next x-tile into LDS (overlapped with GEMM);
// gates = [x;h] @ W^T via v_wmma_f32_16x16x32_bf16; one barrier per step.
// The per-k-block weight offset is laundered (not the pointer!) so the loads
// stay inside the s-loop (no spill-hoisting) while keeping global provenance
// -> global_load_b128 (LOADcnt only), not flat_load (LOADcnt+DScnt).
// ---------------------------------------------------------------------------
__global__ __launch_bounds__(512, 1) void lstm_persist(
    const unsigned short* __restrict__ embBF,
    const unsigned short* __restrict__ Wfrag,
    const float* __restrict__ bias,
    const int* __restrict__ lengths,
    float* __restrict__ out) {
  __shared__ unsigned short Xp[2][16][XSTRIDE] __attribute__((aligned(16)));
  __shared__ unsigned short Hp[2][16][XSTRIDE] __attribute__((aligned(16)));

  const int tid  = threadIdx.x;
  const int wave = tid >> 5, lane = tid & 31;
  const int m0 = lane & 15, half = lane >> 4;
  const int wg = blockIdx.x;
  const int dir = wg >> 2, bb = wg & 3;

  // h0 = 0
  for (int c = tid; c < 16 * (H_ / 8); c += 512) {
    int row = c >> 6, c8 = c & 63;
    *(uint4*)&Hp[0][row][c8 * 8] = make_uint4(0, 0, 0, 0);
  }

#if HAVE_TDM
  {  // prologue: DMA x-tile for step 0 into Xp[0]
    const int xs0 = dir ? (S_ - 1) : 0;
    if (wave == 0)
      issue_tdm_x(embBF + ((size_t)xs0 * B_ + bb * 16) * E_,
                  (unsigned int)(size_t)(const void*)&Xp[0][0][0]);
  }
#endif

  float bias_r[2][4];
#pragma unroll
  for (int q = 0; q < 2; ++q)
#pragma unroll
    for (int g = 0; g < 4; ++g)
      bias_r[q][g] = bias[dir * FOURH + g * H_ + (wave + 16 * q) * 16 + m0];

  int lenm1[8];
#pragma unroll
  for (int r = 0; r < 8; ++r) lenm1[r] = lengths[bb * 16 + r + 8 * half] - 1;

  float cst[2][8];
#pragma unroll
  for (int q = 0; q < 2; ++q)
#pragma unroll
    for (int r = 0; r < 8; ++r) cst[q][r] = 0.0f;

  const unsigned short* Wd = Wfrag + (size_t)dir * FOURH * KTOT;
  // Single weight base: shorts offset = wave*16384 + lane*16; per-tile offsets
  // (g*524288 + q*262144 shorts) fold into imm24 of each load.
  const unsigned short* wb = Wd + (size_t)wave * 16384 + (size_t)lane * 16;

  float* outH = out + (size_t)S_ * B_ * 1024;  // final_h
  float* outC = outH + B_ * H_;                // final_c

  for (int s = 0; s < S_; ++s) {
    const int p = s & 1;

#if HAVE_TDM
    __builtin_amdgcn_s_wait_tensorcnt(0);  // no-op for waves with TENSORcnt==0
    __syncthreads();                       // Xp[p] + Hp[p] ready for all waves
    if (wave == 0 && s + 1 < S_) {         // prefetch next x-tile under the GEMM
      const int xsn = dir ? (S_ - 2 - s) : (s + 1);
      issue_tdm_x(embBF + ((size_t)xsn * B_ + bb * 16) * E_,
                  (unsigned int)(size_t)(const void*)&Xp[p ^ 1][0][0]);
    }
#else
    {  // fallback: manual fill of Xp[p]
      const int xs = dir ? (S_ - 1 - s) : s;
      const unsigned short* src = embBF + ((size_t)xs * B_ + bb * 16) * E_;
      for (int c = tid; c < 16 * (E_ / 8); c += 512) {
        int row = c >> 6, c8 = c & 63;
        *(uint4*)&Xp[p][row][c8 * 8] = *(const uint4*)(src + row * E_ + c8 * 8);
      }
      __syncthreads();
    }
#endif

    v8f acc[2][4];
#pragma unroll
    for (int q = 0; q < 2; ++q)
#pragma unroll
      for (int g = 0; g < 4; ++g)
#pragma unroll
        for (int r = 0; r < 8; ++r) acc[q][g][r] = 0.0f;

    // k-blocks 0..15: x panel
#pragma unroll 2
    for (int kb = 0; kb < 16; ++kb) {
      unsigned int koff = (unsigned int)kb * 512u;   // shorts
      asm volatile("" : "+v"(koff));   // opaque offset: loads stay inside s-loop,
                                       // base keeps global provenance
      const unsigned short* wkb = wb + koff;
      Frag a;   // A in this code's contiguous order: k = 16*half + 0..15 (matches Wfrag)
      a.q[0] = *(const uint4*)&Xp[p][m0][kb * 32 + half * 16];
      a.q[1] = *(const uint4*)&Xp[p][m0][kb * 32 + half * 16 + 8];
#pragma unroll
      for (int q = 0; q < 2; ++q)
#pragma unroll
        for (int g = 0; g < 4; ++g) {
          const uint4* wp =
              (const uint4*)(wkb + (size_t)g * 524288 + (size_t)q * 262144);
          Frag b; b.q[0] = wp[0]; b.q[1] = wp[1];
          acc[q][g] = __builtin_amdgcn_wmma_f32_16x16x32_bf16(
              false, a.v, false, b.v, (short)0, acc[q][g], false, false);
        }
    }
    // k-blocks 16..31: h panel
#pragma unroll 2
    for (int kb = 16; kb < 32; ++kb) {
      unsigned int koff = (unsigned int)kb * 512u;
      asm volatile("" : "+v"(koff));
      const unsigned short* wkb = wb + koff;
      Frag a;
      a.q[0] = *(const uint4*)&Hp[p][m0][(kb - 16) * 32 + half * 16];
      a.q[1] = *(const uint4*)&Hp[p][m0][(kb - 16) * 32 + half * 16 + 8];
#pragma unroll
      for (int q = 0; q < 2; ++q)
#pragma unroll
        for (int g = 0; g < 4; ++g) {
          const uint4* wp =
              (const uint4*)(wkb + (size_t)g * 524288 + (size_t)q * 262144);
          Frag b; b.q[0] = wp[0]; b.q[1] = wp[1];
          acc[q][g] = __builtin_amdgcn_wmma_f32_16x16x32_bf16(
              false, a.v, false, b.v, (short)0, acc[q][g], false, false);
        }
    }

    // LSTM cell elementwise + writeback
    float* orow = out + (size_t)s * (B_ * 1024) + (size_t)(bb * 16) * 1024 + dir * H_;
#pragma unroll
    for (int q = 0; q < 2; ++q) {
      int j = (wave + 16 * q) * 16 + m0;
#pragma unroll
      for (int r = 0; r < 8; ++r) {
        int mrow = r + 8 * half;   // C/D layout: VGPR r -> M = r + 8*(lane/16)
        float iv = acc[q][0][r] + bias_r[q][0];
        float fv = acc[q][1][r] + bias_r[q][1];
        float gv = acc[q][2][r] + bias_r[q][2];
        float ov = acc[q][3][r] + bias_r[q][3];
        float cn = sigm(fv) * cst[q][r] + sigm(iv) * tanh_f(gv);
        float hv = sigm(ov) * tanh_f(cn);
        cst[q][r] = cn;
        Hp[p ^ 1][mrow][j] = f2bf(hv);            // h for next step
        orow[(size_t)mrow * 1024 + j] = hv;       // output (S,B,2H)
        if (dir == 0 && s == lenm1[r]) {          // final states (fwd only)
          outH[(size_t)(bb * 16 + mrow) * H_ + j] = hv;
          outC[(size_t)(bb * 16 + mrow) * H_ + j] = cn;
        }
      }
    }
  }
}

// ---------------------------------------------------------------------------
extern "C" void kernel_launch(void* const* d_in, const int* in_sizes, int n_in,
                              void* d_out, int out_size, void* d_ws, size_t ws_size,
                              hipStream_t stream) {
  const int*   x       = (const int*)  d_in[0];
  const int*   lengths = (const int*)  d_in[1];
  const float* lang    = (const float*)d_in[2];
  const float* table   = (const float*)d_in[3];
  const float* wih_w   = (const float*)d_in[8];
  const float* wih_b   = (const float*)d_in[9];
  const float* whh_w   = (const float*)d_in[10];
  const float* whh_b   = (const float*)d_in[11];
  const float* bih_w   = (const float*)d_in[12];
  const float* bih_b   = (const float*)d_in[13];
  const float* bhh_w   = (const float*)d_in[14];
  const float* bhh_b   = (const float*)d_in[15];
  const float* wih_r_w = (const float*)d_in[16];
  const float* wih_r_b = (const float*)d_in[17];
  const float* whh_r_w = (const float*)d_in[18];
  const float* whh_r_b = (const float*)d_in[19];
  const float* bih_r_w = (const float*)d_in[20];
  const float* bih_r_b = (const float*)d_in[21];
  const float* bhh_r_w = (const float*)d_in[22];
  const float* bhh_r_b = (const float*)d_in[23];
  float* out = (float*)d_out;

  char* ws = (char*)d_ws;
  unsigned short* Wfrag = (unsigned short*)ws;                              // 8 MB
  unsigned short* embBF = (unsigned short*)(ws + (size_t)8  * 1024 * 1024); // 16 MB
  float*          bias  = (float*)         (ws + (size_t)24 * 1024 * 1024); // 16 KB

  hipLaunchKernelGGL(embed_bf16_kernel, dim3(32768), dim3(256), 0, stream,
                     x, table, embBF);
  hipLaunchKernelGGL(genw_kernel, dim3(16384), dim3(256), 0, stream, lang,
                     wih_w, wih_b, whh_w, whh_b,
                     wih_r_w, wih_r_b, whh_r_w, whh_r_b, Wfrag);
  hipLaunchKernelGGL(genb_kernel, dim3(16), dim3(256), 0, stream, lang,
                     bih_w, bih_b, bhh_w, bhh_b,
                     bih_r_w, bih_r_b, bhh_r_w, bhh_r_b, bias);
  hipLaunchKernelGGL(lstm_persist, dim3(8), dim3(512), 0, stream,
                     embBF, Wfrag, bias, lengths, out);
}
